// SARAEngine_10136122818952
// MI455X (gfx1250) — compile-verified, matrix operand order checked by reading, not run on hardware
//
#include <hip/hip_runtime.h>
#include <hip/hip_bf16.h>

// ---------------------------------------------------------------------------
// Types for CDNA5 WMMA
// ---------------------------------------------------------------------------
typedef __attribute__((ext_vector_type(16))) _Float16 v16h;
typedef __attribute__((ext_vector_type(8)))  _Float16 v8h;
typedef __attribute__((ext_vector_type(8)))  float    v8f;

#define B_SZ    2048
#define IN_DIM  4096
#define NENC    1024
#define T_STEPS 20
#define D_STATE 16
#define D_CONV  4
#define D_INNER 4096
#define DT_RANK 64
#define H_MEAN  1024
#define N_OUT   1000
#define PROJ    8288

__device__ __forceinline__ float sigmoidf_(float x) { return 1.0f / (1.0f + __expf(-x)); }
__device__ __forceinline__ float siluf_(float x)    { return x * sigmoidf_(x); }
__device__ __forceinline__ float softplusf_(float x){ return (x > 20.0f) ? x : log1pf(__expf(x)); }

// ---------------------------------------------------------------------------
// f32 -> f16 convert
// ---------------------------------------------------------------------------
__global__ void cvt_f32_f16_kernel(const float* __restrict__ in,
                                   _Float16* __restrict__ out, long n) {
    long i = (long)blockIdx.x * blockDim.x + threadIdx.x;
    if (i < n) out[i] = (_Float16)in[i];
}

// ---------------------------------------------------------------------------
// Generic WMMA GEMM:  C(M,N) = act( A16(M,K; lda) @ B16(N,K; ldb)^T + bias )
// Block tile 128x128, K-step 32, 256 threads = 8 waves, each wave 32x64
// (2x4 grid of 16x16 v_wmma_f32_16x16x32_f16 accumulators).
//
// Staging uses the gfx1250 async DMA path: global_load_async_to_lds_b128
// (ASYNCcnt) with double-buffered LDS, so the global fetch of stage k+1
// overlaps the WMMA compute of stage k. Out-of-range rows are CLAMPED, not
// zero-filled: a garbage A row only feeds output rows >= M, a garbage B row
// only feeds output column n >= N, and both are discarded by the guarded
// epilogue -- so staging is branch-free (EXEC all-1s at every issue site).
//
// act: 0 = none, 1 = softplus.  C32 / C16 each optional (nullable).
// ---------------------------------------------------------------------------
#define LDS_STRIDE 56   // halves per LDS row (112 B: 16B-aligned, conflict-free)

__global__ __launch_bounds__(256, 2)
void wmma_gemm_kernel(const _Float16* __restrict__ A, long lda, long M, long K,
                      const _Float16* __restrict__ Bw, long ldb, long N,
                      const float* __restrict__ bias,
                      float* __restrict__ C32, _Float16* __restrict__ C16,
                      int act) {
    __shared__ _Float16 sA[2][128 * LDS_STRIDE];
    __shared__ _Float16 sB[2][128 * LDS_STRIDE];

    const int tid  = threadIdx.x;
    const int wv   = tid >> 5;       // wave id 0..7
    const int lane = tid & 31;
    const int wm   = wv & 3;         // wave M position (0..3) -> 32 rows each
    const int wn   = wv >> 2;        // wave N position (0..1) -> 64 cols each
    const int hh   = lane >> 4;      // lane half (K-phase selector)
    const int mm   = lane & 15;      // row/col within 16

    const long bm = (long)blockIdx.y * 128;
    const long bn = (long)blockIdx.x * 128;

    v8f acc[2][4];
#pragma unroll
    for (int i = 0; i < 2; ++i)
#pragma unroll
        for (int j = 0; j < 4; ++j)
#pragma unroll
            for (int r = 0; r < 8; ++r) acc[i][j][r] = 0.0f;

    // staging coordinates: 4 threads per 32-half row, 64 rows/pass, 2 passes
    const int tr = tid >> 2;          // 0..63
    const int tc = (tid & 3) << 3;    // 0,8,16,24 (halves)

    // clamped global rows (safe: see header comment)
    long gaRow[2], gbRow[2];
#pragma unroll
    for (int p = 0; p < 2; ++p) {
        long ga = bm + p * 64 + tr; gaRow[p] = (ga < M) ? ga : (M - 1);
        long gb = bn + p * 64 + tr; gbRow[p] = (gb < N) ? gb : (N - 1);
    }

    // async-issue one 128x32 A tile + 128x32 B tile into LDS buffer `buf`
    auto issue_stage = [&](int buf, long kb) {
#pragma unroll
        for (int p = 0; p < 2; ++p) {
            const int srow = (p * 64 + tr) * LDS_STRIDE + tc;
            unsigned la = (unsigned)(uintptr_t)(&sA[buf][srow]);
            unsigned ga = (unsigned)((gaRow[p] * lda + kb + tc) * 2);
            asm volatile("global_load_async_to_lds_b128 %0, %1, %2"
                         :: "v"(la), "v"(ga), "s"(A) : "memory");
            unsigned lb = (unsigned)(uintptr_t)(&sB[buf][srow]);
            unsigned gb = (unsigned)((gbRow[p] * ldb + kb + tc) * 2);
            asm volatile("global_load_async_to_lds_b128 %0, %1, %2"
                         :: "v"(lb), "v"(gb), "s"(Bw) : "memory");
        }
    };

    int buf = 0;
    issue_stage(0, 0);                               // prologue: 4 async ops in flight

    for (long kb = 0; kb < K; kb += 32) {
        if (kb + 32 < K) {
            issue_stage(buf ^ 1, kb + 32);           // 8 in flight
            // async ops retire in order: <=4 means current stage has landed
            asm volatile("s_wait_asynccnt 0x4" ::: "memory");
        } else {
            asm volatile("s_wait_asynccnt 0x0" ::: "memory");
        }
        __syncthreads();                             // stage `buf` visible to all waves

        // ---- load fragments per documented 16-bit A/B VGPR layout ----
        // lane holds row (mm); halves [8h..8h+7] and [16+8h..16+8h+7]
        v16h af[2], bf[4];
#pragma unroll
        for (int mt = 0; mt < 2; ++mt) {
            const int row = wm * 32 + mt * 16 + mm;
            const v8h* pa = (const v8h*)(&sA[buf][row * LDS_STRIDE]);
            v8h lo = pa[hh];
            v8h hi = pa[2 + hh];
            af[mt] = __builtin_shufflevector(lo, hi, 0,1,2,3,4,5,6,7,8,9,10,11,12,13,14,15);
        }
#pragma unroll
        for (int nt = 0; nt < 4; ++nt) {
            const int row = wn * 64 + nt * 16 + mm;
            const v8h* pb = (const v8h*)(&sB[buf][row * LDS_STRIDE]);
            v8h lo = pb[hh];
            v8h hi = pb[2 + hh];
            bf[nt] = __builtin_shufflevector(lo, hi, 0,1,2,3,4,5,6,7,8,9,10,11,12,13,14,15);
        }

        // ---- 8 WMMAs per K-step per wave ----
#pragma unroll
        for (int mt = 0; mt < 2; ++mt)
#pragma unroll
            for (int nt = 0; nt < 4; ++nt)
                acc[mt][nt] = __builtin_amdgcn_wmma_f32_16x16x32_f16(
                    false, af[mt], false, bf[nt], (short)0, acc[mt][nt], false, false);

        __syncthreads();   // all reads of `buf` done before it is overwritten
        buf ^= 1;
    }

    // ---- epilogue: 16x16 C tile layout (VGPR r: M=r / M=8+r; lanes = N) ----
#pragma unroll
    for (int mt = 0; mt < 2; ++mt) {
#pragma unroll
        for (int nt = 0; nt < 4; ++nt) {
            const long col = bn + wn * 64 + nt * 16 + mm;
            if (col >= N) continue;
            const float bv = bias ? bias[col] : 0.0f;
#pragma unroll
            for (int r = 0; r < 8; ++r) {
                const long row = bm + wm * 32 + mt * 16 + hh * 8 + r;
                if (row >= M) continue;
                float v = acc[mt][nt][r] + bv;
                if (act == 1) v = softplusf_(v);
                if (C32) C32[row * N + col] = v;
                if (C16) C16[row * N + col] = (_Float16)v;
            }
        }
    }
}

// ---------------------------------------------------------------------------
// LayerNorm: one block (256 thr) per row; optional f32 and f16 outputs
// ---------------------------------------------------------------------------
__global__ __launch_bounds__(256)
void layernorm_kernel(const float* __restrict__ in, int rowLen,
                      const float* __restrict__ g, const float* __restrict__ b,
                      float* __restrict__ out32, _Float16* __restrict__ out16) {
    __shared__ float rs[256], rs2[256];
    const int tid = threadIdx.x;
    const float* p = in + (size_t)blockIdx.x * rowLen;
    float s = 0.0f, s2 = 0.0f;
    for (int i = tid; i < rowLen; i += 256) { float v = p[i]; s += v; s2 += v * v; }
    rs[tid] = s; rs2[tid] = s2; __syncthreads();
    for (int st = 128; st > 0; st >>= 1) {
        if (tid < st) { rs[tid] += rs[tid + st]; rs2[tid] += rs2[tid + st]; }
        __syncthreads();
    }
    const float mu  = rs[0] / rowLen;
    const float var = rs2[0] / rowLen - mu * mu;
    const float rstd = rsqrtf(var + 1e-5f);
    for (int i = tid; i < rowLen; i += 256) {
        float v = (p[i] - mu) * rstd * g[i] + b[i];
        if (out32) out32[(size_t)blockIdx.x * rowLen + i] = v;
        if (out16) out16[(size_t)blockIdx.x * rowLen + i] = (_Float16)v;
    }
}

// ---------------------------------------------------------------------------
// LIF: 20-step membrane loop in registers; spikes -> f16 (binary, exact);
// spike-count partial sums atomically folded into rate (pre-scaled).
// ---------------------------------------------------------------------------
__global__ __launch_bounds__(256)
void lif_kernel(const float* __restrict__ cur, _Float16* __restrict__ spikes,
                float* __restrict__ rate_acc) {
    __shared__ float red[256];
    const long idx = (long)blockIdx.x * 256 + threadIdx.x;   // b*1024 + n
    const long b = idx >> 10;
    const long n = idx & 1023;
    const float c = cur[idx];
    float v = 0.0f, cnt = 0.0f;
#pragma unroll
    for (int t = 0; t < T_STEPS; ++t) {
        v = 0.5f * (v + c);                        // v += (cur - v)/TAU, TAU=2
        const float s = (v - 1.0f > 0.0f) ? 1.0f : 0.0f;
        v -= s;
        spikes[((b * T_STEPS + t) << 10) + n] = (_Float16)s;
        cnt += s;
    }
    red[threadIdx.x] = cnt; __syncthreads();
    for (int st = 128; st > 0; st >>= 1) {
        if (threadIdx.x < st) red[threadIdx.x] += red[threadIdx.x + st];
        __syncthreads();
    }
    if (threadIdx.x == 0)
        atomicAdd(rate_acc, red[0] * (1.0f / ((float)B_SZ * T_STEPS * NENC)));
}

// ---------------------------------------------------------------------------
// Depthwise causal conv (k=4) over t + SiLU:  xin16 (B*T, D_INNER) -> xc16
// ---------------------------------------------------------------------------
__global__ __launch_bounds__(256)
void conv_silu_kernel(const _Float16* __restrict__ xin,
                      const float* __restrict__ cw, const float* __restrict__ cb,
                      _Float16* __restrict__ xc) {
    const long idx = (long)blockIdx.x * 256 + threadIdx.x;   // (b*T+t)*4096 + c
    const long c  = idx & (D_INNER - 1);
    const long bt = idx >> 12;
    const long b  = bt / T_STEPS;
    const long t  = bt % T_STEPS;
    float acc = cb[c];
#pragma unroll
    for (int k = 0; k < D_CONV; ++k) {
        const long tt = t + k - (D_CONV - 1);
        if (tt >= 0)
            acc += (float)xin[((b * T_STEPS + tt) << 12) + c] * cw[c * D_CONV + k];
    }
    xc[idx] = (_Float16)siluf_(acc);
}

// ---------------------------------------------------------------------------
// Selective-scan: per (b, channel) thread, h[16] in registers, 20 steps.
// Emits only t=19 output, fused with +x*D and *silu(z).
// bcdt layout per (b,t) row of 96 f32: [B(16) | C(16) | dt_r(64)].
// ---------------------------------------------------------------------------
__global__ __launch_bounds__(256)
void ssm_scan_kernel(const _Float16* __restrict__ dt16,
                     const _Float16* __restrict__ xc16,
                     const float* __restrict__ bcdt,
                     const float* __restrict__ A_log,
                     const float* __restrict__ Dp,
                     const float* __restrict__ z32,
                     float* __restrict__ fm) {
    const long idx = (long)blockIdx.x * 256 + threadIdx.x;   // b*4096 + c
    const long c = idx & (D_INNER - 1);
    const long b = idx >> 12;

    float Ar[D_STATE];
#pragma unroll
    for (int s = 0; s < D_STATE; ++s) Ar[s] = -__expf(A_log[c * D_STATE + s]);

    float h[D_STATE];
#pragma unroll
    for (int s = 0; s < D_STATE; ++s) h[s] = 0.0f;

    float y = 0.0f, xlast = 0.0f;
#pragma unroll
    for (int t = 0; t < T_STEPS; ++t) {
        const long row = b * T_STEPS + t;
        const float dt = (float)dt16[(row << 12) + c];
        const float x  = (float)xc16[(row << 12) + c];
        const float* bc = bcdt + row * 96;                   // wave-uniform -> broadcast
#pragma unroll
        for (int s = 0; s < D_STATE; ++s)
            h[s] = __expf(Ar[s] * dt) * h[s] + dt * bc[s] * x;
        if (t == T_STEPS - 1) {
#pragma unroll
            for (int s = 0; s < D_STATE; ++s) y += h[s] * bc[16 + s];
            xlast = x;
        }
    }
    y += xlast * Dp[c];
    y *= siluf_(z32[(b << 12) + c]);
    fm[(b << 12) + c] = y;
}

// ---------------------------------------------------------------------------
// GRU gate fusion: gi/gh (B, 3*H) -> h (B, H) update, f32 + f16 copies
// ---------------------------------------------------------------------------
__global__ __launch_bounds__(256)
void gru_gate_kernel(const float* __restrict__ gi, const float* __restrict__ gh,
                     float* __restrict__ h32, _Float16* __restrict__ h16) {
    const long idx = (long)blockIdx.x * 256 + threadIdx.x;   // b*1024 + j
    const long b = idx >> 10;
    const long j = idx & (H_MEAN - 1);
    const long o = b * 3 * H_MEAN + j;
    const float r  = sigmoidf_(gi[o]              + gh[o]);
    const float zz = sigmoidf_(gi[o + H_MEAN]     + gh[o + H_MEAN]);
    const float n  = tanhf    (gi[o + 2 * H_MEAN] + r * gh[o + 2 * H_MEAN]);
    const float hn = (1.0f - zz) * n + zz * h32[idx];
    h32[idx] = hn;
    h16[idx] = (_Float16)hn;
}

// ---------------------------------------------------------------------------
// Host orchestration
// ---------------------------------------------------------------------------
extern "C" void kernel_launch(void* const* d_in, const int* in_sizes, int n_in,
                              void* d_out, int out_size, void* d_ws, size_t ws_size,
                              hipStream_t stream) {
    const float* x       = (const float*)d_in[0];
    const float* fc_w    = (const float*)d_in[1];
    const float* fc_b    = (const float*)d_in[2];
    const float* enc_g   = (const float*)d_in[3];
    const float* enc_bt  = (const float*)d_in[4];
    const float* in_w    = (const float*)d_in[5];
    const float* in_b    = (const float*)d_in[6];
    const float* conv_w  = (const float*)d_in[7];
    const float* conv_b  = (const float*)d_in[8];
    const float* dtp_w   = (const float*)d_in[9];
    const float* dtp_b   = (const float*)d_in[10];
    const float* A_log   = (const float*)d_in[11];
    const float* Dp      = (const float*)d_in[12];
    const float* out_w   = (const float*)d_in[13];
    const float* out_b   = (const float*)d_in[14];
    const float* mnorm_g = (const float*)d_in[15];
    const float* mnorm_b = (const float*)d_in[16];
    const float* w_ih    = (const float*)d_in[17];
    const float* w_hh    = (const float*)d_in[18];
    const float* b_ih    = (const float*)d_in[19];
    const float* b_hh    = (const float*)d_in[20];
    const float* rln_g   = (const float*)d_in[21];
    const float* rln_b   = (const float*)d_in[22];
    const float* dec_w   = (const float*)d_in[23];
    const float* dec_b   = (const float*)d_in[24];

    float* out  = (float*)d_out;
    float* rate = out + (size_t)B_SZ * N_OUT;

    // ---- workspace carve (bump allocator, 256B aligned) ----
    char* wsb = (char*)d_ws; size_t off = 0;
    auto take = [&](size_t bytes) -> void* {
        void* p = wsb + off;
        off += (bytes + 255) & ~(size_t)255;
        return p;
    };
    const long BT = (long)B_SZ * T_STEPS;
    _Float16* x16     = (_Float16*)take((size_t)B_SZ * IN_DIM * 2);
    _Float16* fcw16   = (_Float16*)take((size_t)NENC * IN_DIM * 2);
    _Float16* inw16   = (_Float16*)take((size_t)PROJ * NENC * 2);
    _Float16* dtpw16  = (_Float16*)take((size_t)D_INNER * DT_RANK * 2);
    _Float16* outw16  = (_Float16*)take((size_t)NENC * D_INNER * 2);
    _Float16* wih16   = (_Float16*)take((size_t)3 * H_MEAN * NENC * 2);
    _Float16* whh16   = (_Float16*)take((size_t)3 * H_MEAN * H_MEAN * 2);
    _Float16* decw16  = (_Float16*)take((size_t)N_OUT * H_MEAN * 2);
    float*    tmp1    = (float*)   take((size_t)B_SZ * NENC * 4);
    float*    cur     = (float*)   take((size_t)B_SZ * NENC * 4);
    _Float16* spikes  = (_Float16*)take((size_t)BT * NENC * 2);
    _Float16* xin16   = (_Float16*)take((size_t)BT * D_INNER * 2);
    _Float16* xc16    = (_Float16*)take((size_t)BT * D_INNER * 2);
    _Float16* dt16    = (_Float16*)take((size_t)BT * D_INNER * 2);
    float*    bcdt32  = (float*)   take((size_t)BT * 96 * 4);
    _Float16* bcdt16  = (_Float16*)take((size_t)BT * 96 * 2);
    float*    z32     = (float*)   take((size_t)B_SZ * D_INNER * 4);
    float*    fm32    = (float*)   take((size_t)B_SZ * D_INNER * 4);
    _Float16* ln16    = (_Float16*)take((size_t)B_SZ * D_INNER * 2);
    _Float16* m16     = (_Float16*)take((size_t)B_SZ * NENC * 2);
    float*    gi32    = (float*)   take((size_t)B_SZ * 3 * H_MEAN * 4);
    float*    gh32    = (float*)   take((size_t)B_SZ * 3 * H_MEAN * 4);
    float*    h32     = (float*)   take((size_t)B_SZ * H_MEAN * 4);
    _Float16* h16     = (_Float16*)take((size_t)B_SZ * H_MEAN * 2);
    _Float16* zf16    = (_Float16*)take((size_t)B_SZ * H_MEAN * 2);

    auto cvt = [&](const float* src, _Float16* dst, long n) {
        cvt_f32_f16_kernel<<<(unsigned)((n + 255) / 256), 256, 0, stream>>>(src, dst, n);
    };
    auto gemm = [&](const _Float16* A, long lda, long M, long K,
                    const _Float16* Bw, long ldb, long N, const float* bias,
                    float* C32, _Float16* C16, int act) {
        dim3 g((unsigned)((N + 127) / 128), (unsigned)((M + 127) / 128));
        wmma_gemm_kernel<<<g, 256, 0, stream>>>(A, lda, M, K, Bw, ldb, N, bias, C32, C16, act);
    };

    // ---- weight / input converts to f16 ----
    cvt(x,     x16,    (long)B_SZ * IN_DIM);
    cvt(fc_w,  fcw16,  (long)NENC * IN_DIM);
    cvt(in_w,  inw16,  (long)PROJ * NENC);
    cvt(dtp_w, dtpw16, (long)D_INNER * DT_RANK);
    cvt(out_w, outw16, (long)NENC * D_INNER);
    cvt(w_ih,  wih16,  (long)3 * H_MEAN * NENC);
    cvt(w_hh,  whh16,  (long)3 * H_MEAN * H_MEAN);
    cvt(dec_w, decw16, (long)N_OUT * H_MEAN);

    // ---- encoder: x @ fc_w^T + fc_b -> layernorm -> cur ----
    gemm(x16, IN_DIM, B_SZ, IN_DIM, fcw16, IN_DIM, NENC, fc_b, tmp1, nullptr, 0);
    layernorm_kernel<<<B_SZ, 256, 0, stream>>>(tmp1, NENC, enc_g, enc_bt, cur, nullptr);

    // ---- LIF spiking (binary -> f16 exact) + rate ----
    hipMemsetAsync(rate, 0, sizeof(float), stream);
    lif_kernel<<<(B_SZ * NENC) / 256, 256, 0, stream>>>(cur, spikes, rate);

    // ---- projections (z pruned to last timestep only) ----
    gemm(spikes, NENC, BT, NENC, inw16 + (size_t)D_INNER * NENC, NENC, D_INNER,
         in_b + D_INNER, nullptr, xin16, 0);                           // x_in, all t
    gemm(spikes, NENC, BT, NENC, inw16 + (size_t)2 * D_INNER * NENC, NENC, 96,
         in_b + 2 * D_INNER, bcdt32, bcdt16, 0);                       // B|C|dt_r, all t
    gemm(spikes + (size_t)(T_STEPS - 1) * NENC, (long)T_STEPS * NENC, B_SZ, NENC,
         inw16, NENC, D_INNER, in_b, z32, nullptr, 0);                 // z, t = T-1 only

    // ---- depthwise causal conv + SiLU ----
    conv_silu_kernel<<<(unsigned)((BT * D_INNER) / 256), 256, 0, stream>>>(
        xin16, conv_w, conv_b, xc16);

    // ---- dt = softplus(dt_r @ dtp_w^T + dtp_b) ----
    gemm(bcdt16 + 32, 96, BT, DT_RANK, dtpw16, DT_RANK, D_INNER, dtp_b,
         nullptr, dt16, 1);

    // ---- selective scan (fused +x*D, *silu(z)) -> final_mem ----
    ssm_scan_kernel<<<(B_SZ * D_INNER) / 256, 256, 0, stream>>>(
        dt16, xc16, bcdt32, A_log, Dp, z32, fm32);

    // ---- mnorm layernorm -> f16, then m = ln @ out_w^T + out_b ----
    layernorm_kernel<<<B_SZ, 256, 0, stream>>>(fm32, D_INNER, mnorm_g, mnorm_b,
                                               nullptr, ln16);
    gemm(ln16, D_INNER, B_SZ, D_INNER, outw16, D_INNER, NENC, out_b,
         nullptr, m16, 0);

    // ---- GRU: gi once, then 5 x (gh GEMM + gate fusion) ----
    gemm(m16, NENC, B_SZ, NENC, wih16, NENC, 3 * H_MEAN, b_ih, gi32, nullptr, 0);
    hipMemsetAsync(h32, 0, (size_t)B_SZ * H_MEAN * 4, stream);
    hipMemsetAsync(h16, 0, (size_t)B_SZ * H_MEAN * 2, stream);
    for (int d = 0; d < 5; ++d) {
        gemm(h16, H_MEAN, B_SZ, H_MEAN, whh16, H_MEAN, 3 * H_MEAN, b_hh,
             gh32, nullptr, 0);
        gru_gate_kernel<<<(B_SZ * H_MEAN) / 256, 256, 0, stream>>>(gi32, gh32, h32, h16);
    }

    // ---- final layernorm + decoder -> logits (directly into d_out) ----
    layernorm_kernel<<<B_SZ, 256, 0, stream>>>(h32, H_MEAN, rln_g, rln_b,
                                               nullptr, zf16);
    gemm(zf16, H_MEAN, B_SZ, H_MEAN, decw16, H_MEAN, N_OUT, dec_b,
         out, nullptr, 0);
}
